// Discriminator_86998857548156
// MI455X (gfx1250) — matrix operand order repaired and checked by reading, not compiled, not run on hardware
//
#include <hip/hip_runtime.h>
#include <hip/hip_bf16.h>

typedef __attribute__((ext_vector_type(16))) __bf16 v16bf;
typedef __attribute__((ext_vector_type(8)))  float  v8f;

#define HH   128
#define BB   512
#define NN   10000

__device__ __forceinline__ float sigm(float v) { return 1.0f / (1.0f + __expf(-v)); }

// ---------------------------------------------------------------------------
// One-time f32 -> bf16 conversion of the WMMA-consumed weights.
// grid 5000, block 256
// ---------------------------------------------------------------------------
__global__ __launch_bounds__(256) void cvt_kernel(
    const float* __restrict__ Wih, const float* __restrict__ Whh,
    const float* __restrict__ Wnode,
    __bf16* __restrict__ wih_bf, __bf16* __restrict__ whh_bf,
    __bf16* __restrict__ wnode_bf)
{
  long i = (long)blockIdx.x * 256 + threadIdx.x;
  if (i < 7L*512*HH) {
    wih_bf[i] = (__bf16)Wih[i];
    whh_bf[i] = (__bf16)Whh[i];
  }
  if (i < (long)HH*NN) wnode_bf[i] = (__bf16)Wnode[i];
}

// ---------------------------------------------------------------------------
// Small embedding projections (K = 1 or 2): pure VALU; outputs bf16 since
// they are consumed only as WMMA A-fragments.   grid (512, 51), block 128
// ---------------------------------------------------------------------------
__global__ __launch_bounds__(128) void embed_kernel(
    const float* __restrict__ x, const float* __restrict__ t0,
    const float* __restrict__ tau, const float* __restrict__ endv,
    const float* __restrict__ coords,
    const float* __restrict__ Wx1, const float* __restrict__ Wx2, const float* __restrict__ bx2,
    const float* __restrict__ Wres, const float* __restrict__ bres,
    const float* __restrict__ Wtau, const float* __restrict__ btau,
    const float* __restrict__ Wend1, const float* __restrict__ Wend2, const float* __restrict__ bend2,
    const float* __restrict__ Wcoord,
    __bf16* __restrict__ xh, __bf16* __restrict__ t0h, __bf16* __restrict__ tauh,
    __bf16* __restrict__ endh, __bf16* __restrict__ coordh)
{
  const int b = blockIdx.x;
  const int r = blockIdx.y;
  const int j = threadIdx.x;
  if (r == 0) {
    float s = x[b*2+0]*Wx1[0] + x[b*2+1]*Wx1[1];
    xh[b*HH + j] = (__bf16)tanhf(s * Wx2[j] + bx2[j]);
  } else if (r == 1) {
    t0h[b*HH + j] = (__bf16)tanhf(t0[b] * Wres[j] + bres[j]);
  } else if (r < 18) {
    int g = r - 2;
    tauh[(b*16 + g)*HH + j] = (__bf16)tanhf(tau[b*16 + g] * Wtau[j] + btau[j]);
  } else if (r == 18) {
    float s = endv[b*2+0]*Wend1[0] + endv[b*2+1]*Wend1[1];
    endh[b*HH + j] = (__bf16)(s * Wend2[j] + bend2[j]);
  } else {
    int tt = r - 19;
    float c0 = coords[(b*32 + tt)*2 + 0];
    float c1 = coords[(b*32 + tt)*2 + 1];
    coordh[(b*32 + tt)*HH + j] = (__bf16)(c0 * Wcoord[j*2+0] + c1 * Wcoord[j*2+1]);
  }
}

// ---------------------------------------------------------------------------
// nodeh = node_inputs(16384 x 10000) @ Wnode.T(10000 x 128)   [HBM-bound]
// One wave per 16x16 output tile; block = 8 waves = one row-tile, 8 col-tiles.
// A is streamed f32 (input dtype) and converted; B is pre-converted bf16
// (L2-resident, 2.56 MB).  Tail is branch-free (valid K range is exactly the
// lower half of each lane's fragment).   grid 1024, block 256
// ---------------------------------------------------------------------------
__global__ __launch_bounds__(256) void node_gemm_kernel(
    const float* __restrict__ A, const __bf16* __restrict__ Wb,
    __bf16* __restrict__ C)
{
  const int lane  = threadIdx.x & 31;
  const int wave  = threadIdx.x >> 5;     // col tile 0..7
  const int mlane = lane & 15;
  const int ksel  = lane >> 4;
  const int rowTile = blockIdx.x;         // 0..1023

  const float*  __restrict__ aRow = A  + (long)(rowTile*16 + mlane) * NN;
  const __bf16* __restrict__ wRow = Wb + (long)(wave*16 + mlane) * NN;

  v8f acc = {};
  int k0 = 0;
  for (; k0 + 32 <= NN; k0 += 32) {
    __builtin_prefetch(aRow + k0 + 512, 0, 0);   // global_prefetch_b8
    v16bf a, b;
#pragma unroll
    for (int i = 0; i < 8; ++i) {
      a[i]   = (__bf16)aRow[k0 + ksel*8 + i];
      a[8+i] = (__bf16)aRow[k0 + 16 + ksel*8 + i];
    }
#pragma unroll
    for (int i = 0; i < 16; ++i)
      b[i] = wRow[k0 + ((i & 8) << 1) + ksel*8 + (i & 7)];
    acc = __builtin_amdgcn_wmma_f32_16x16x32_bf16(false, a, false, b, (short)0, acc, false, false);
  }
  { // K tail: k0 = 9984, exactly 16 valid columns -> branch-free
    const __bf16 z = (__bf16)0.0f;
    v16bf a, b;
#pragma unroll
    for (int i = 0; i < 8; ++i) {
      a[i]   = (__bf16)aRow[k0 + ksel*8 + i];  // 9984..9999: always in range
      a[8+i] = z;                              // K=10000.. : zero
    }
#pragma unroll
    for (int i = 0; i < 16; ++i) {
      __bf16 v = wRow[k0 + ksel*8 + (i & 7)];  // in range for every lane
      b[i] = (i & 8) ? z : v;                  // upper 8 elements cover K>=10000
    }
    acc = __builtin_amdgcn_wmma_f32_16x16x32_bf16(false, a, false, b, (short)0, acc, false, false);
  }
#pragma unroll
  for (int r = 0; r < 8; ++r) {
    int row = rowTile*16 + r + 8*ksel;
    int col = wave*16 + mlane;
    C[(long)row*HH + col] = (__bf16)acc[r];
  }
}

// ---------------------------------------------------------------------------
// Sequence-element gather map: (lstm k, step t) -> row base + batch stride
// ---------------------------------------------------------------------------
__device__ __forceinline__ void seq_src(int k, int t,
    const __bf16* xh, const __bf16* t0h, const __bf16* tauh,
    const __bf16* nodeh, const __bf16* coordh, const __bf16* endh,
    const __bf16*& src, int& bstride)
{
  switch (k) {
    case 0: {
      if (t == 0)       { src = xh;   bstride = HH; }
      else if (t == 1)  { src = t0h;  bstride = HH; }
      else if (t == 82) { src = endh; bstride = HH; }
      else {
        int i = t - 2, g = i / 5, p = i % 5;
        if (p == 0)      { src = tauh   + g*HH;        bstride = 16*HH; }
        else if (p == 1) { src = nodeh  + (2*g)*HH;    bstride = 32*HH; }
        else if (p == 2) { src = coordh + (2*g)*HH;    bstride = 32*HH; }
        else if (p == 3) { src = nodeh  + (2*g+1)*HH;  bstride = 32*HH; }
        else             { src = coordh + (2*g+1)*HH;  bstride = 32*HH; }
      }
    } break;
    case 1: {
      if (t == 0)       { src = xh;   bstride = HH; }
      else if (t == 1)  { src = t0h;  bstride = HH; }
      else if (t == 18) { src = endh; bstride = HH; }
      else              { src = tauh + (t-2)*HH; bstride = 16*HH; }
    } break;
    case 2: { src = nodeh  + t*HH; bstride = 32*HH; } break;
    case 3: { src = coordh + t*HH; bstride = 32*HH; } break;
    case 4: {
      if (t == 0)       { src = xh;   bstride = HH; }
      else if (t == 1)  { src = t0h;  bstride = HH; }
      else if (t == 50) { src = endh; bstride = HH; }
      else {
        int i = t - 2, g = i / 3, p = i % 3;
        if (p == 0)      { src = tauh  + g*HH;        bstride = 16*HH; }
        else if (p == 1) { src = nodeh + (2*g)*HH;    bstride = 32*HH; }
        else             { src = nodeh + (2*g+1)*HH;  bstride = 32*HH; }
      }
    } break;
    case 5: {
      int g = t / 4, p = t % 4;
      if (p == 0)      { src = nodeh  + (2*g)*HH;   }
      else if (p == 1) { src = coordh + (2*g)*HH;   }
      else if (p == 2) { src = nodeh  + (2*g+1)*HH; }
      else             { src = coordh + (2*g+1)*HH; }
      bstride = 32*HH;
    } break;
    default: {
      if (t == 0)       { src = xh;   bstride = HH; }
      else if (t == 1)  { src = t0h;  bstride = HH; }
      else if (t == 50) { src = endh; bstride = HH; }
      else {
        int i = t - 2, g = i / 3, p = i % 3;
        if (p == 0)      { src = tauh   + g*HH;        bstride = 16*HH; }
        else if (p == 1) { src = coordh + (2*g)*HH;    bstride = 32*HH; }
        else             { src = coordh + (2*g+1)*HH;  bstride = 32*HH; }
      }
    } break;
  }
}

// ---------------------------------------------------------------------------
// Persistent LSTM: one workgroup per (batch slice of 32, lstm id).
// Both Wih and Whh cached bf16 in LDS (256 KB) + h (bf16) + c (f32) + bias
// => 282 KB of the 320 KB WGP LDS, one workgroup per WGP.  Each wave owns the
// i/f/g/o tiles of two 16x16 (batch x hidden) blocks, so the cell update stays
// in registers; only the x_t gather touches global.  grid (16, 7), block 256
// ---------------------------------------------------------------------------
__global__ __launch_bounds__(256) void lstm_kernel(
    const __bf16* __restrict__ wih_bf, const __bf16* __restrict__ whh_bf,
    const float* __restrict__ bih, const float* __restrict__ bhh,
    const __bf16* __restrict__ xh, const __bf16* __restrict__ t0h,
    const __bf16* __restrict__ tauh, const __bf16* __restrict__ nodeh,
    const __bf16* __restrict__ coordh, const __bf16* __restrict__ endh,
    float* __restrict__ hfinal)
{
  const int k    = blockIdx.y;
  const int b0   = blockIdx.x * 32;
  const int tid  = threadIdx.x;
  const int lane = tid & 31;
  const int wave = tid >> 5;
  const int mlane = lane & 15;
  const int ksel  = lane >> 4;

  __shared__ __bf16 wihS[512*HH];   // 128 KB
  __shared__ __bf16 whhS[512*HH];   // 128 KB
  __shared__ __bf16 hS[32*HH];      //   8 KB
  __shared__ float  cS[32*HH];      //  16 KB
  __shared__ float  biasS[512];     //   2 KB

  { // stage weights (dword copies, already bf16) and init state
    const uint* si = (const uint*)(wih_bf + (long)k * 512 * HH);
    const uint* sh = (const uint*)(whh_bf + (long)k * 512 * HH);
    uint* di = (uint*)wihS;
    uint* dh = (uint*)whhS;
    for (int i = tid; i < 512*HH/2; i += 256) { di[i] = si[i]; dh[i] = sh[i]; }
    const float* bihK = bih + k * 512;
    const float* bhhK = bhh + k * 512;
    for (int i = tid; i < 512; i += 256) biasS[i] = bihK[i] + bhhK[i];
    for (int i = tid; i < 32*HH; i += 256) { hS[i] = (__bf16)0.0f; cS[i] = 0.0f; }
  }
  __syncthreads();

  const int Ts[7] = {83, 19, 32, 32, 51, 64, 51};
  const int T = Ts[k];

  for (int t = 0; t < T; ++t) {
    const __bf16* src; int bstride;
    seq_src(k, t, xh, t0h, tauh, nodeh, coordh, endh, src, bstride);

    v8f acc[2][4];
#pragma unroll
    for (int grp = 0; grp < 2; ++grp)
#pragma unroll
      for (int g = 0; g < 4; ++g) { v8f z = {}; acc[grp][g] = z; }

#pragma unroll
    for (int grp = 0; grp < 2; ++grp) {
      const int ggi = wave + grp*8;      // 0..15
      const int mt  = ggi >> 3;          // batch sub-tile 0..1
      const int jt  = ggi & 7;           // hidden tile 0..7

      // recurrent: h @ Whh^T   (A and B both from LDS bf16)
#pragma unroll
      for (int kk = 0; kk < HH; kk += 32) {
        v16bf a;
        const int hrow = (mt*16 + mlane) * HH;
#pragma unroll
        for (int i = 0; i < 8; ++i) {
          a[i]   = hS[hrow + kk + ksel*8 + i];
          a[8+i] = hS[hrow + kk + 16 + ksel*8 + i];
        }
#pragma unroll
        for (int g = 0; g < 4; ++g) {
          const int n = g*HH + jt*16 + mlane;
          v16bf bf;
#pragma unroll
          for (int i = 0; i < 16; ++i)
            bf[i] = whhS[n*HH + kk + ((i & 8) << 1) + ksel*8 + (i & 7)];
          acc[grp][g] = __builtin_amdgcn_wmma_f32_16x16x32_bf16(
              false, a, false, bf, (short)0, acc[grp][g], false, false);
        }
      }

      // input: x_t @ Wih^T   (A gathered bf16 from global, B from LDS bf16)
      const __bf16* xrow = src + (long)(b0 + mt*16 + mlane) * bstride;
#pragma unroll
      for (int kk = 0; kk < HH; kk += 32) {
        v16bf a;
#pragma unroll
        for (int i = 0; i < 8; ++i) {
          a[i]   = xrow[kk + ksel*8 + i];
          a[8+i] = xrow[kk + 16 + ksel*8 + i];
        }
#pragma unroll
        for (int g = 0; g < 4; ++g) {
          const int n = g*HH + jt*16 + mlane;
          v16bf bf;
#pragma unroll
          for (int i = 0; i < 16; ++i)
            bf[i] = wihS[n*HH + kk + ((i & 8) << 1) + ksel*8 + (i & 7)];
          acc[grp][g] = __builtin_amdgcn_wmma_f32_16x16x32_bf16(
              false, a, false, bf, (short)0, acc[grp][g], false, false);
        }
      }
    }
    __syncthreads();   // all GEMM reads of hS done before we overwrite it

    // cell update, fully in registers (C/D layout: N=lane&15, M=r+8*ksel)
#pragma unroll
    for (int grp = 0; grp < 2; ++grp) {
      const int ggi = wave + grp*8;
      const int mt  = ggi >> 3;
      const int jt  = ggi & 7;
#pragma unroll
      for (int r = 0; r < 8; ++r) {
        const int ml = mt*16 + r + 8*ksel;
        const int j  = jt*16 + mlane;
        const float iv = acc[grp][0][r] + biasS[0*HH + j];
        const float fv = acc[grp][1][r] + biasS[1*HH + j];
        const float gv = acc[grp][2][r] + biasS[2*HH + j];
        const float ov = acc[grp][3][r] + biasS[3*HH + j];
        const float co = cS[ml*HH + j];
        const float cn = sigm(fv) * co + sigm(iv) * tanhf(gv);
        cS[ml*HH + j] = cn;
        hS[ml*HH + j] = (__bf16)(sigm(ov) * tanhf(cn));
      }
    }
    __syncthreads();
  }

  for (int i = tid; i < 32*HH; i += 256) {
    const int ml = i >> 7, j = i & 127;
    hfinal[((long)k*BB + b0 + ml)*HH + j] = (float)hS[i];
  }
}

// ---------------------------------------------------------------------------
// out[k][b] = tanh(h @ W1^T + b1) @ W2^T + b2   — tiny, VALU.
// grid (512, 7), block 128
// ---------------------------------------------------------------------------
__global__ __launch_bounds__(128) void head_kernel(
    const float* __restrict__ hfinal, const float* __restrict__ W1,
    const float* __restrict__ b1, const float* __restrict__ W2,
    const float* __restrict__ b2, float* __restrict__ out)
{
  const int b = blockIdx.x, k = blockIdx.y, j = threadIdx.x;
  const float* hv = hfinal + ((long)k*BB + b) * HH;
  const float* w  = W1 + ((long)k*HH + j) * HH;
  float s = 0.f;
#pragma unroll 4
  for (int i = 0; i < HH; ++i) s += hv[i] * w[i];
  float hu = tanhf(s + b1[k*HH + j]) * W2[k*HH + j];
  __shared__ float red[128];
  red[j] = hu;
  __syncthreads();
  for (int off = 64; off > 0; off >>= 1) {
    if (j < off) red[j] += red[j + off];
    __syncthreads();
  }
  if (j == 0) out[k*BB + b] = red[0] + b2[k];
}

// ---------------------------------------------------------------------------
extern "C" void kernel_launch(void* const* d_in, const int* in_sizes, int n_in,
                              void* d_out, int out_size, void* d_ws, size_t ws_size,
                              hipStream_t stream)
{
  const float* x        = (const float*)d_in[0];
  const float* t0_res   = (const float*)d_in[1];
  const float* node_in  = (const float*)d_in[2];
  const float* tau_in   = (const float*)d_in[3];
  const float* endv     = (const float*)d_in[4];
  const float* coords   = (const float*)d_in[5];
  const float* Wx1      = (const float*)d_in[6];
  const float* Wx2      = (const float*)d_in[7];
  const float* bx2      = (const float*)d_in[8];
  const float* Wres     = (const float*)d_in[9];
  const float* bres     = (const float*)d_in[10];
  const float* Wnode    = (const float*)d_in[11];
  const float* Wtau     = (const float*)d_in[12];
  const float* btau     = (const float*)d_in[13];
  const float* Wend1    = (const float*)d_in[14];
  const float* Wend2    = (const float*)d_in[15];
  const float* bend2    = (const float*)d_in[16];
  const float* Wcoord   = (const float*)d_in[17];
  const float* lstm_Wih = (const float*)d_in[18];
  const float* lstm_Whh = (const float*)d_in[19];
  const float* lstm_bih = (const float*)d_in[20];
  const float* lstm_bhh = (const float*)d_in[21];
  const float* head_W1  = (const float*)d_in[22];
  const float* head_b1  = (const float*)d_in[23];
  const float* head_W2  = (const float*)d_in[24];
  const float* head_b2  = (const float*)d_in[25];

  // workspace carve-up (bf16 activations/weights, f32 hfinal)
  __bf16* nodeh    = (__bf16*)d_ws;            // 2,097,152
  __bf16* coordh   = nodeh  + 2097152;         // 2,097,152
  __bf16* xh       = coordh + 2097152;         //    65,536
  __bf16* t0h      = xh     + 65536;           //    65,536
  __bf16* tauh     = t0h    + 65536;           // 1,048,576
  __bf16* endh     = tauh   + 1048576;         //    65,536
  __bf16* wih_bf   = endh   + 65536;           //   458,752
  __bf16* whh_bf   = wih_bf + 458752;          //   458,752
  __bf16* wnode_bf = whh_bf + 458752;          // 1,280,000
  float*  hfin     = (float*)(wnode_bf + 1280000); // 458,752 f32 (~17 MB total)

  cvt_kernel<<<dim3(5000), 256, 0, stream>>>(
      lstm_Wih, lstm_Whh, Wnode, wih_bf, whh_bf, wnode_bf);

  embed_kernel<<<dim3(512, 51), 128, 0, stream>>>(
      x, t0_res, tau_in, endv, coords,
      Wx1, Wx2, bx2, Wres, bres, Wtau, btau, Wend1, Wend2, bend2, Wcoord,
      xh, t0h, tauh, endh, coordh);

  node_gemm_kernel<<<dim3(1024), 256, 0, stream>>>(node_in, wnode_bf, nodeh);

  lstm_kernel<<<dim3(16, 7), 256, 0, stream>>>(
      wih_bf, whh_bf, lstm_bih, lstm_bhh,
      xh, t0h, tauh, nodeh, coordh, endh, hfin);

  head_kernel<<<dim3(512, 7), 128, 0, stream>>>(
      hfin, head_W1, head_b1, head_W2, head_b2, (float*)d_out);
}
